// Model_7868380086797
// MI455X (gfx1250) — matrix operand order repaired and checked by reading, not compiled, not run on hardware
//
#include <hip/hip_runtime.h>

typedef __attribute__((ext_vector_type(16))) __bf16 v16bf;
typedef __attribute__((ext_vector_type(8)))  float  v8f;

#define NB  16
#define CC  64
#define TT  64
#define VV  1024
#define KNN 20
#define OC  64
#define IC  128  // 2*C

#define NEG_INF (-3.402823466e38f)

// ---- workspace layout (bytes), all offsets 256B aligned ----
#define OFF_XMV   ((size_t)0)         // f32  [N][V][C]   4 MB
#define OFF_XMH   ((size_t)4194304)   // bf16 [N][V][C]   2 MB
#define OFF_XX    ((size_t)6291456)   // f32  [N][V]      64 KB
#define OFF_WH    ((size_t)6356992)   // bf16 [O][128]    16 KB
#define OFF_SCALE ((size_t)6373376)   // f32  [64]
#define OFF_SHIFT ((size_t)6373632)   // f32  [64]
#define OFF_IDX   ((size_t)6373888)   // i32  [N][V][K]   1.25 MB
#define OFF_HOUT  ((size_t)7684608)   // f32  [N][O][V]   4 MB
#define OFF_DIST  ((size_t)11878912)  // f32  [N][V][V]   64 MB

// ---------------- 1) mean over T ----------------
// x: [N,C,T,V] -> xmv f32 [N,V,C], xmh bf16 [N,V,C]
__global__ void mean_kernel(const float* __restrict__ x,
                            float* __restrict__ xmv,
                            __bf16* __restrict__ xmh) {
    int tid = blockIdx.x * blockDim.x + threadIdx.x;  // v fastest -> coalesced x reads
    int v = tid & (VV - 1);
    int c = (tid >> 10) & (CC - 1);
    int n = tid >> 16;
    const float* p = x + ((size_t)(n * CC + c) * TT) * VV + v;
    float s = 0.f;
#pragma unroll 8
    for (int t = 0; t < TT; ++t) s += p[(size_t)t * VV];
    s *= (1.0f / TT);
    size_t o = (size_t)(n * VV + v) * CC + c;
    xmv[o] = s;
    xmh[o] = (__bf16)s;
}

// ---------------- 2) xx, bf16 weights, folded BN ----------------
__global__ void prep_kernel(const float* __restrict__ xmv,
                            const float* __restrict__ conv_w,
                            const float* __restrict__ gamma,
                            const float* __restrict__ beta,
                            const float* __restrict__ rmean,
                            const float* __restrict__ rvar,
                            float* __restrict__ xx,
                            __bf16* __restrict__ wh,
                            float* __restrict__ scale,
                            float* __restrict__ shift) {
    int tid = blockIdx.x * blockDim.x + threadIdx.x;
    if (tid < NB * VV) {
        const float* p = xmv + (size_t)tid * CC;
        float s = 0.f;
#pragma unroll 8
        for (int c = 0; c < CC; ++c) s += p[c] * p[c];
        xx[tid] = s;
    }
    if (tid < OC * IC) wh[tid] = (__bf16)conv_w[tid];
    if (tid < OC) {
        float inv = gamma[tid] * rsqrtf(rvar[tid] + 1e-5f);
        scale[tid] = inv;
        shift[tid] = beta[tid] - rmean[tid] * inv;
    }
}

// ---------------- 3) Gram/distance via bf16 WMMA ----------------
// one wave per 16x16 tile of D[n] = Xm^T * Xm  (K = C = 64 -> 2 WMMAs)
__global__ void dist_kernel(const __bf16* __restrict__ xmh,
                            const float* __restrict__ xx,
                            float* __restrict__ dist) {
    int wave = blockIdx.x * 8 + (threadIdx.x >> 5);
    int lane = threadIdx.x & 31;
    int n  = wave >> 12;
    int vt = (wave >> 6) & 63;
    int ut = wave & 63;
    int m    = lane & 15;
    int half = lane >> 4;

    const __bf16* arow = xmh + (size_t)(n * VV + vt * 16 + m) * CC;  // A row = v
    const __bf16* brow = xmh + (size_t)(n * VV + ut * 16 + m) * CC;  // B col = u

    v8f acc = {};
#pragma unroll
    for (int chunk = 0; chunk < 2; ++chunk) {
        v16bf a, b;
#pragma unroll
        for (int e = 0; e < 16; ++e) {
            // A 16x32 bf16 layout: k = 16*(e>>3) + 8*half + (e&7)
            a[e] = arow[chunk * 32 + ((e >> 3) << 4) + (half << 3) + (e & 7)];
            // B 32x16 bf16 layout: k = 16*(e>>3) + 8*half + (e&7)
            b[e] = brow[chunk * 32 + ((e >> 3) << 4) + (half << 3) + (e & 7)];
        }
        acc = __builtin_amdgcn_wmma_f32_16x16x32_bf16(false, a, false, b,
                                                      (short)0, acc, false, false);
    }
    int u = ut * 16 + m;
    float xxu = xx[n * VV + u];
#pragma unroll
    for (int j = 0; j < 8; ++j) {
        int vr = vt * 16 + half * 8 + j;                 // C/D layout: M = 8*half + j
        float d = 2.0f * acc[j] - xx[n * VV + vr] - xxu; // -xx_v + 2*inner - xx_u
        dist[(size_t)(n * VV + vr) * VV + u] = d;
    }
}

// ---------------- 4) top-k=20 per row, one wave per row ----------------
__global__ void topk_kernel(const float* __restrict__ dist, int* __restrict__ idx) {
    int row  = blockIdx.x * 8 + (threadIdx.x >> 5);  // n*V + v
    int lane = threadIdx.x & 31;
    const float* r = dist + (size_t)row * VV;
    float vals[32];
#pragma unroll
    for (int j = 0; j < 32; ++j) vals[j] = r[lane + 32 * j];

    for (int kk = 0; kk < KNN; ++kk) {
        float bv = NEG_INF; int bu = 0x7fffffff;
#pragma unroll
        for (int j = 0; j < 32; ++j) {
            int u = lane + 32 * j;
            bool take = (vals[j] > bv) || (vals[j] == bv && u < bu);
            bv = take ? vals[j] : bv;
            bu = take ? u : bu;
        }
#pragma unroll
        for (int off = 16; off > 0; off >>= 1) {
            float ov = __shfl_xor(bv, off, 32);
            int   ou = __shfl_xor(bu, off, 32);
            bool take = (ov > bv) || (ov == bv && ou < bu);
            bv = take ? ov : bv;
            bu = take ? ou : bu;
        }
        if (lane == 0) idx[(size_t)row * KNN + kk] = bu;
        if ((bu & 31) == lane) {
            int jj = bu >> 5;
#pragma unroll
            for (int j = 0; j < 32; ++j)
                if (j == jj) vals[j] = NEG_INF;  // mask winner, stays in regs
        }
    }
}

// ---------------- 5) gather + 1x1 conv (bf16 WMMA) + BN + Leaky + max_k ----------------
// block = 128 threads (4 waves); block handles (n, 16-v tile); wave w owns o in [16w,16w+16)
__global__ void __launch_bounds__(128)
edge_kernel(const float* __restrict__ xmv, const __bf16* __restrict__ wh,
            const int* __restrict__ idx, const float* __restrict__ scale,
            const float* __restrict__ shift, float* __restrict__ hout) {
    __shared__ __bf16 F[10 * 16 * IC];  // 40 KB: feat[kk][m][c] for 10 neighbors
    int n  = blockIdx.x >> 6;
    int vt = blockIdx.x & 63;
    int lane = threadIdx.x & 31;
    int w    = threadIdx.x >> 5;
    int m    = lane & 15;
    int half = lane >> 4;
    int o    = w * 16 + m;  // output channel column for this lane

    // weights resident in VGPRs: B[k][o] = conv_w[o][k]
    v16bf bw[4];
#pragma unroll
    for (int chunk = 0; chunk < 4; ++chunk)
#pragma unroll
        for (int e = 0; e < 16; ++e)
            bw[chunk][e] = wh[o * IC + chunk * 32 + ((e >> 3) << 4) + (half << 3) + (e & 7)];

    float sc = scale[o], sh = shift[o];
    float hmax[8];
#pragma unroll
    for (int j = 0; j < 8; ++j) hmax[j] = NEG_INF;

    for (int kh = 0; kh < 2; ++kh) {
        // cooperative gather of feat tile: (nbr - ctr | ctr), bf16
        for (int id = threadIdx.x; id < 10 * 16 * IC; id += 128) {
            int c  = id & (IC - 1);
            int mm = (id >> 7) & 15;
            int kk = id >> 11;  // 0..9
            int v = vt * 16 + mm;
            int u = idx[(size_t)(n * VV + v) * KNN + kh * 10 + kk];
            const float* ctr = xmv + (size_t)(n * VV + v) * CC;
            const float* nbr = xmv + (size_t)(n * VV + u) * CC;
            float val = (c < CC) ? (nbr[c] - ctr[c]) : ctr[c - CC];
            F[id] = (__bf16)val;
        }
        __syncthreads();

        for (int kk = 0; kk < 10; ++kk) {
            v8f acc = {};
#pragma unroll
            for (int chunk = 0; chunk < 4; ++chunk) {  // K = 128 -> 4 WMMAs
                v16bf a;
#pragma unroll
                for (int e = 0; e < 16; ++e)
                    a[e] = F[(kk * 16 + m) * IC + chunk * 32 +
                             ((e >> 3) << 4) + (half << 3) + (e & 7)];
                acc = __builtin_amdgcn_wmma_f32_16x16x32_bf16(false, a, false, bw[chunk],
                                                              (short)0, acc, false, false);
            }
#pragma unroll
            for (int j = 0; j < 8; ++j) {
                float y = acc[j] * sc + sh;
                y = (y >= 0.f) ? y : 0.2f * y;         // LeakyReLU(0.2)
                hmax[j] = fmaxf(hmax[j], y);           // max over neighbors
            }
        }
        __syncthreads();
    }
#pragma unroll
    for (int j = 0; j < 8; ++j) {
        int vr = vt * 16 + half * 8 + j;
        hout[(size_t)(n * OC + o) * VV + vr] = hmax[j];
    }
}

// ---------------- 6) broadcast over T (coalesced 256 MB store) ----------------
__global__ void bcast_kernel(const float* __restrict__ hout, float* __restrict__ out) {
    size_t tid = (size_t)blockIdx.x * blockDim.x + threadIdx.x;  // [N,O,T,V], v fastest
    int v = (int)(tid & (VV - 1));
    int o = (int)((tid >> 16) & (OC - 1));
    int n = (int)(tid >> 22);
    out[tid] = hout[(size_t)(n * OC + o) * VV + v];
}

extern "C" void kernel_launch(void* const* d_in, const int* in_sizes, int n_in,
                              void* d_out, int out_size, void* d_ws, size_t ws_size,
                              hipStream_t stream) {
    const float* x      = (const float*)d_in[0];
    const float* conv_w = (const float*)d_in[1];
    const float* gamma  = (const float*)d_in[2];
    const float* beta   = (const float*)d_in[3];
    const float* rmean  = (const float*)d_in[4];
    const float* rvar   = (const float*)d_in[5];

    char* ws = (char*)d_ws;
    float*  xmv   = (float*)(ws + OFF_XMV);
    __bf16* xmh   = (__bf16*)(ws + OFF_XMH);
    float*  xx    = (float*)(ws + OFF_XX);
    __bf16* wh    = (__bf16*)(ws + OFF_WH);
    float*  scale = (float*)(ws + OFF_SCALE);
    float*  shift = (float*)(ws + OFF_SHIFT);
    int*    idx   = (int*)(ws + OFF_IDX);
    float*  hout  = (float*)(ws + OFF_HOUT);
    float*  dist  = (float*)(ws + OFF_DIST);
    float*  out   = (float*)d_out;

    mean_kernel<<<(NB * CC * VV) / 256, 256, 0, stream>>>(x, xmv, xmh);
    prep_kernel<<<(NB * VV) / 256, 256, 0, stream>>>(xmv, conv_w, gamma, beta,
                                                     rmean, rvar, xx, wh, scale, shift);
    dist_kernel<<<(NB * 64 * 64) / 8, 256, 0, stream>>>(xmh, xx, dist);
    topk_kernel<<<(NB * VV) / 8, 256, 0, stream>>>(dist, idx);
    edge_kernel<<<NB * 64, 128, 0, stream>>>(xmv, wh, idx, scale, shift, hout);
    bcast_kernel<<<(NB * OC * TT * VV) / 256, 256, 0, stream>>>(hout, out);
}
